// RelGraphConvLayer_81793357185090
// MI455X (gfx1250) — matrix operand order repaired and checked by reading, not compiled, not run on hardware
//
#include <hip/hip_runtime.h>
#include <stdint.h>

typedef __attribute__((ext_vector_type(16))) _Float16 v16h;
typedef __attribute__((ext_vector_type(8)))  _Float16 v8h;
typedef __attribute__((ext_vector_type(8)))  float    v8f;

#define HID 128          // hidden dim (fixed by reference)
#define NREL 8           // relation count (fixed by reference)

// ---- gfx1250 async global->LDS copy (ASYNCcnt tracked, bypasses VGPRs) ------
__device__ __forceinline__ void async_g2l_b128(uint32_t lds_off, const void* gptr) {
    asm volatile("global_load_async_to_lds_b128 %0, %1, off"
                 :: "v"(lds_off), "v"((uint64_t)(uintptr_t)gptr) : "memory");
}
__device__ __forceinline__ void wait_async0() {
    asm volatile("s_wait_asynccnt 0x0" ::: "memory");
}

// ---------------------------------------------------------------- f32 -> f16
__global__ void k_f2h(const float* __restrict__ x, _Float16* __restrict__ y, long n) {
    long i = (long)blockIdx.x * blockDim.x + threadIdx.x;
    if (i < n) y[i] = (_Float16)x[i];
}

// --------------- per-relation in-degree + bucket counts (LDS histogram) ------
__global__ void __launch_bounds__(256)
k_count(const int* __restrict__ etype, const int* __restrict__ dstA,
        float* __restrict__ deg, int* __restrict__ counts, int E, int N) {
    __shared__ int hist[NREL];
    const int t = threadIdx.x;
    if (t < NREL) hist[t] = 0;
    __syncthreads();
    int e = blockIdx.x * 256 + t;
    if (e < E) {
        int r = etype[e] & (NREL - 1);
        atomicAdd(&hist[r], 1);                                   // ds_add
        atomicAdd(&deg[(size_t)r * N + dstA[e]], 1.0f);           // scattered
    }
    __syncthreads();
    if (t < NREL && hist[t] > 0) atomicAdd(&counts[t], hist[t]);  // <=8/block
}

// ------------------------------------------------ tiny exclusive scan (1 thread)
__global__ void k_offsets(const int* __restrict__ counts, int* __restrict__ boff,
                          int* __restrict__ goff, int* __restrict__ cursor) {
    int off = 0, go = 0;
    for (int r = 0; r < NREL; ++r) {
        boff[r] = off; goff[r] = go; cursor[r] = off;
        off += counts[r];
        go  += (counts[r] + 31) >> 5;     // 32-edge groups
    }
    boff[NREL] = off; goff[NREL] = go;
}

// ---------------- scatter edge ids into buckets (block-aggregated atomics) ---
__global__ void __launch_bounds__(256)
k_scatter(const int* __restrict__ etype, int* __restrict__ cursor,
          int* __restrict__ perm, int E) {
    __shared__ int hist[NREL];
    __shared__ int base[NREL];
    const int t = threadIdx.x;
    if (t < NREL) hist[t] = 0;
    __syncthreads();
    int e = blockIdx.x * 256 + t;
    int r = 0, rank = 0;
    if (e < E) {
        r = etype[e] & (NREL - 1);
        rank = atomicAdd(&hist[r], 1);    // ds_add_rtn -> local rank
    }
    __syncthreads();
    if (t < NREL) base[t] = hist[t] ? atomicAdd(&cursor[t], hist[t]) : 0;
    __syncthreads();
    if (e < E) perm[base[r] + rank] = e;
}

// ---------------- self GEMM: out = h @ W_self^T + b -------------------------
// 32-row band per block; 8 waves cover 128 cols. Each wave keeps TWO f32
// accumulators (independent WMMA chains, co-executable) and reuses each B
// fragment for both row bands. K=128 -> 4 k-steps x 2 wmma.
__global__ void __launch_bounds__(256)
k_self_gemm(const _Float16* __restrict__ h16, const _Float16* __restrict__ w16,
            const float* __restrict__ bias, float* __restrict__ out, int N) {
    const int row0 = blockIdx.x * 32;
    const int wave = threadIdx.x >> 5;
    const int lane = threadIdx.x & 31;
    const int lm   = lane & 15;
    const int hi   = lane >> 4;
    const int col  = wave * 16 + lm;

    int m0 = row0 + lm;      if (m0 >= N) m0 = N - 1;
    int m1 = row0 + 16 + lm; if (m1 >= N) m1 = N - 1;
    const _Float16* a0row = h16 + (size_t)m0 * HID;
    const _Float16* a1row = h16 + (size_t)m1 * HID;
    const _Float16* brow  = w16 + (size_t)col * HID;    // B[k][col] = W[col][k]

    v8f acc0 = {}, acc1 = {};
#pragma unroll
    for (int kk = 0; kk < 4; ++kk) {
        v8h b0 = *(const v8h*)(brow + kk * 32 + hi * 8), b1 = *(const v8h*)(brow + kk * 32 + hi * 8 + 16);
        v16h b = __builtin_shufflevector(b0, b1, 0, 1, 2, 3, 4, 5, 6, 7, 8, 9, 10, 11, 12, 13, 14, 15);
        const _Float16* p0 = a0row + kk * 32 + hi * 8;
        const _Float16* p1 = a1row + kk * 32 + hi * 8;
        v8h x0 = *(const v8h*)(p0), x1 = *(const v8h*)(p0 + 16);
        v8h y0 = *(const v8h*)(p1), y1 = *(const v8h*)(p1 + 16);
        v16h a0, a1;
#pragma unroll
        for (int i = 0; i < 8; ++i) {
            a0[i] = x0[i]; a0[8 + i] = x1[i];
            a1[i] = y0[i]; a1[8 + i] = y1[i];
        }
        acc0 = __builtin_amdgcn_wmma_f32_16x16x32_f16(false, a0, false, b,
                                                      (short)0, acc0, false, false);
        acc1 = __builtin_amdgcn_wmma_f32_16x16x32_f16(false, a1, false, b,
                                                      (short)0, acc1, false, false);
    }
    const float bv = bias[col];
    if (row0 + 32 <= N) {                 // uniform fast path: branchless stores
#pragma unroll
        for (int r = 0; r < 8; ++r) {
            int m = row0 + r + 8 * hi;
            out[(size_t)m * HID + col]        = acc0[r] + bv;
            out[(size_t)(m + 16) * HID + col] = acc1[r] + bv;
        }
    } else {
#pragma unroll
        for (int r = 0; r < 8; ++r) {
            int m = row0 + r + 8 * hi;
            if (m < N)      out[(size_t)m * HID + col]        = acc0[r] + bv;
            if (m + 16 < N) out[(size_t)(m + 16) * HID + col] = acc1[r] + bv;
        }
    }
}

// ---------------- edge message GEMM ------------------------------------------
// One block = one 32-edge group of a single relation bucket. The 32 gathered
// h_src rows (8 KB) are staged in LDS via gfx1250 async global->LDS copies,
// then 8 waves x 2 accumulators run the f16 WMMAs; results are scaled by
// 1/max(deg,1) and atomically accumulated into out[dst] (L2-resident).
__global__ void __launch_bounds__(256)
k_edge_gemm(const _Float16* __restrict__ h16, const _Float16* __restrict__ wrel16,
            const int* __restrict__ srcA, const int* __restrict__ dstA,
            const int* __restrict__ perm, const int* __restrict__ boff,
            const int* __restrict__ goff, const float* __restrict__ deg,
            float* __restrict__ out, int N) {
    __shared__ _Float16 sA[32 * HID];     // 32 gathered rows (f16) = 8 KB
    __shared__ int   ssrc[32];
    __shared__ int   sdst[32];
    __shared__ float sscale[32];

    const int b = blockIdx.x;
    if (b >= goff[NREL]) return;          // uniform: past last group
    int rb = 0;
#pragma unroll
    for (int r = 1; r < NREL; ++r) if (b >= goff[r]) rb = r;   // bucket search

    const int base = boff[rb] + (b - goff[rb]) * 32;
    const int bend = boff[rb + 1];
    const int t    = threadIdx.x;

    if (t < 32) {
        int gidx = base + t;
        int sv = 0, dv = -1; float sc = 0.f;
        if (gidx < bend) {
            int e = perm[gidx];
            sv = srcA[e];
            dv = dstA[e];
            sc = 1.0f / fmaxf(deg[(size_t)rb * N + dv], 1.0f);
        }
        ssrc[t] = sv; sdst[t] = dv; sscale[t] = sc;
    }
    __syncthreads();

    {   // async gather: 32 rows x 256 B; 256 threads x 32 B (2 x b128 each)
        int row = t >> 3, chunk = t & 7;            // chunk = 32 B
        const _Float16* g = h16 + (size_t)ssrc[row] * HID + chunk * 16;
        uint32_t l = (uint32_t)(uintptr_t)&sA[row * HID + chunk * 16];
        async_g2l_b128(l,      g);
        async_g2l_b128(l + 16, g + 8);
    }
    wait_async0();
    __syncthreads();

    const int wave = t >> 5;
    const int lane = t & 31;
    const int lm   = lane & 15;
    const int hi   = lane >> 4;
    const int col  = wave * 16 + lm;
    const _Float16* wr = wrel16 + (size_t)rb * HID * HID + (size_t)col * HID;

    v8f acc0 = {}, acc1 = {};
#pragma unroll
    for (int kk = 0; kk < 4; ++kk) {
        v8h bf0 = *(const v8h*)(wr + kk * 32 + hi * 8), bf1 = *(const v8h*)(wr + kk * 32 + hi * 8 + 16);
        v16h bf = __builtin_shufflevector(bf0, bf1, 0, 1, 2, 3, 4, 5, 6, 7, 8, 9, 10, 11, 12, 13, 14, 15);
        const _Float16* p0 = &sA[lm * HID + kk * 32 + hi * 8];          // rows 0..15
        const _Float16* p1 = &sA[(16 + lm) * HID + kk * 32 + hi * 8];   // rows 16..31
        v8h x0 = *(const v8h*)(p0), x1 = *(const v8h*)(p0 + 16);
        v8h y0 = *(const v8h*)(p1), y1 = *(const v8h*)(p1 + 16);
        v16h a0, a1;
#pragma unroll
        for (int i = 0; i < 8; ++i) {
            a0[i] = x0[i]; a0[8 + i] = x1[i];
            a1[i] = y0[i]; a1[8 + i] = y1[i];
        }
        acc0 = __builtin_amdgcn_wmma_f32_16x16x32_f16(false, a0, false, bf,
                                                      (short)0, acc0, false, false);
        acc1 = __builtin_amdgcn_wmma_f32_16x16x32_f16(false, a1, false, bf,
                                                      (short)0, acc1, false, false);
    }
#pragma unroll
    for (int r = 0; r < 8; ++r) {
        int m0 = r + 8 * hi;              // D-layout row for band 0
        int d0 = sdst[m0];
        if (d0 >= 0) atomicAdd(&out[(size_t)d0 * HID + col], acc0[r] * sscale[m0]);
        int m1 = 16 + m0;                 // band 1
        int d1 = sdst[m1];
        if (d1 >= 0) atomicAdd(&out[(size_t)d1 * HID + col], acc1[r] * sscale[m1]);
    }
}

// ---------------------------------------- ReLU + LayerNorm (one wave32 per row)
__global__ void __launch_bounds__(128)
k_relu_ln(float* __restrict__ out, const float* __restrict__ gamma,
          const float* __restrict__ beta, int N) {
    int row = blockIdx.x * 4 + (threadIdx.x >> 5);
    if (row >= N) return;
    int lane = threadIdx.x & 31;
    float x[4];
    float s = 0.f;
#pragma unroll
    for (int i = 0; i < 4; ++i) {
        float v = out[(size_t)row * HID + i * 32 + lane];
        v = fmaxf(v, 0.f);
        x[i] = v; s += v;
    }
#pragma unroll
    for (int m = 16; m >= 1; m >>= 1) s += __shfl_xor(s, m, 32);
    float mu = s * (1.0f / HID);
    float vs = 0.f;
#pragma unroll
    for (int i = 0; i < 4; ++i) { float d = x[i] - mu; vs += d * d; }
#pragma unroll
    for (int m = 16; m >= 1; m >>= 1) vs += __shfl_xor(vs, m, 32);
    float inv = rsqrtf(vs * (1.0f / HID) + 1e-5f);
#pragma unroll
    for (int i = 0; i < 4; ++i) {
        int c = i * 32 + lane;
        out[(size_t)row * HID + c] = (x[i] - mu) * inv * gamma[c] + beta[c];
    }
}

// ------------------------------------------------------------------- launcher
extern "C" void kernel_launch(void* const* d_in, const int* in_sizes, int n_in,
                              void* d_out, int out_size, void* d_ws, size_t ws_size,
                              hipStream_t stream) {
    const float* h     = (const float*)d_in[0];
    const int*   eidx  = (const int*)d_in[1];   // [2,E]: row0=src, row1=dst
    const int*   etype = (const int*)d_in[2];
    const float* Wself = (const float*)d_in[3];
    const float* bself = (const float*)d_in[4];
    const float* Wrel  = (const float*)d_in[5];
    const float* gamma = (const float*)d_in[6];
    const float* beta  = (const float*)d_in[7];

    const int N = in_sizes[0] / HID;
    const int E = in_sizes[2];
    const int* srcA = eidx;
    const int* dstA = eidx + E;

    // ---- workspace carve (256B aligned): ~32 MB total
    char* ws = (char*)d_ws;
    size_t o = 0;
    auto take = [&](size_t bytes) { char* p = ws + o; o += (bytes + 255) & ~(size_t)255; return p; };
    _Float16* h16   = (_Float16*)take((size_t)N * HID * sizeof(_Float16));
    _Float16* ws16  = (_Float16*)take((size_t)HID * HID * sizeof(_Float16));
    _Float16* wr16  = (_Float16*)take((size_t)NREL * HID * HID * sizeof(_Float16));
    float*    deg   = (float*)take((size_t)NREL * N * sizeof(float));
    int*      perm  = (int*)take((size_t)E * sizeof(int));
    int*      counts= (int*)take(NREL * sizeof(int));
    int*      boff  = (int*)take((NREL + 1) * sizeof(int));
    int*      goff  = (int*)take((NREL + 1) * sizeof(int));
    int*      cursor= (int*)take(NREL * sizeof(int));
    float*    out   = (float*)d_out;

    hipMemsetAsync(deg, 0, (size_t)NREL * N * sizeof(float), stream);
    hipMemsetAsync(counts, 0, NREL * sizeof(int), stream);

    // f32 -> f16 staging (h16 is L2-resident afterwards for the edge gathers)
    long nh = (long)N * HID;
    k_f2h<<<(int)((nh + 255) / 256), 256, 0, stream>>>(h, h16, nh);
    k_f2h<<<(HID * HID + 255) / 256, 256, 0, stream>>>(Wself, ws16, HID * HID);
    k_f2h<<<(NREL * HID * HID + 255) / 256, 256, 0, stream>>>(Wrel, wr16, NREL * HID * HID);

    // counting sort of edges by relation + per-relation in-degree
    k_count  <<<(E + 255) / 256, 256, 0, stream>>>(etype, dstA, deg, counts, E, N);
    k_offsets<<<1, 1, 0, stream>>>(counts, boff, goff, cursor);
    k_scatter<<<(E + 255) / 256, 256, 0, stream>>>(etype, cursor, perm, E);

    // self term (fully overwrites d_out -> no stale state across graph replays)
    k_self_gemm<<<(N + 31) / 32, 256, 0, stream>>>(h16, ws16, bself, out, N);

    // relation-bucketed edge messages (upper bound on group count for the grid)
    int maxGroups = (E + 31) / 32 + NREL;
    k_edge_gemm<<<maxGroups, 256, 0, stream>>>(h16, wr16, srcA, dstA, perm,
                                               boff, goff, deg, out, N);

    // fused ReLU + LayerNorm, in place
    k_relu_ln<<<(N + 3) / 4, 128, 0, stream>>>(out, gamma, beta, N);
}
